// SATLayer_regular_65000035058134
// MI455X (gfx1250) — matrix operand order repaired and checked by reading, not compiled, not run on hardware
//
#include <hip/hip_runtime.h>
#include <hip/hip_bf16.h>

typedef __attribute__((ext_vector_type(16))) _Float16 v16h;
typedef __attribute__((ext_vector_type(8)))  float    v8f;

#define D    128
#define PADW 136   // LDS row stride (halves) to stagger banks

__device__ __forceinline__ float sigmoidf_(float x) {
    return 1.0f / (1.0f + __expf(-x));
}

// ---------------------------------------------------------------------------
// H = act(X @ W + b)  [N,128] fp32, W staged to LDS as f16 (transposed).
// Optionally fused scalar projection: avec[n] = H[n,:] @ aw + ab.
// One wave = one 16x128 output tile; 8 waves/block = 128 rows/block.
// v_wmma_f32_16x16x32_f16, K split in 4 steps of 32, N in 8 tiles of 16.
// Compile-time LEAKY / STORE / AVEC so unused paths are eliminated.
// ---------------------------------------------------------------------------
template <bool LEAKY, bool STORE, bool AVEC>
__global__ void __launch_bounds__(256)
gemm128_kernel(const float* __restrict__ X, const float* __restrict__ W,
               const float* __restrict__ bias,
               const float* __restrict__ aw, const float* __restrict__ ab,
               float* __restrict__ H, float* __restrict__ avec, int N)
{
    __shared__ _Float16 sW[D * PADW];   // sW[n*PADW + k] = W[k][n] (transposed)
    __shared__ float    sBias[D];
    __shared__ float    sAw[D];

    const int tid  = threadIdx.x;
    const int wave = tid >> 5;
    const int lane = tid & 31;
    const int half = lane >> 4;
    const int l16  = lane & 15;

    // stage weights (fp32 -> f16, transposed), bias, projection vector
    for (int idx = tid; idx < D * D; idx += 256) {
        const int k = idx >> 7;        // row of W
        const int n = idx & (D - 1);   // col of W
        sW[n * PADW + k] = (_Float16)W[idx];
    }
    if (tid < D) {
        sBias[tid] = bias[tid];
        if (AVEC) sAw[tid] = aw[tid];
    }
    __syncthreads();

    const int r0 = blockIdx.x * 128 + wave * 16;
    if (r0 >= N) return;               // wave-uniform
    const bool fullTile = (r0 + 16 <= N);   // wave-uniform

    // A fragments: f16 16x32 layout, k = 16*half + i, row = l16
    const int m  = r0 + l16;
    const int mc = (m < N) ? m : (N - 1);
    const float* xr = X + (size_t)mc * D;
    v16h afrag[4];
    #pragma unroll
    for (int kt = 0; kt < 4; ++kt) {
        const int kb = kt * 32;
        #pragma unroll
        for (int i = 0; i < 16; ++i) {
            const int k = kb + 16 * half + i;
            afrag[kt][i] = (_Float16)xr[k];
        }
    }

    float apart[8];
    #pragma unroll
    for (int r = 0; r < 8; ++r) apart[r] = 0.0f;
    const float abv = AVEC ? ab[0] : 0.0f;

    #pragma unroll
    for (int nt = 0; nt < 8; ++nt) {
        v8f acc = {};
        #pragma unroll
        for (int kt = 0; kt < 4; ++kt) {
            const int kb = kt * 32;
            // B fragment: column n = nt*16+l16, k = kb + 16*half + i (contiguous)
            const _Float16* wp = &sW[(nt * 16 + l16) * PADW + kb + 16 * half];
            v16h bfrag;
            #pragma unroll
            for (int i = 0; i < 16; ++i) bfrag[i] = wp[i];
            acc = __builtin_amdgcn_wmma_f32_16x16x32_f16(
                false, afrag[kt], false, bfrag, (short)0, acc, false, false);
        }
        // epilogue: C/D vgpr r -> row r0 + r + 8*half, col nt*16 + l16
        const int col = nt * 16 + l16;
        float vv[8];
        #pragma unroll
        for (int r = 0; r < 8; ++r) {
            float v = acc[r] + sBias[col];
            if (LEAKY) v = (v >= 0.0f) ? v : 0.2f * v;
            vv[r] = v;
        }
        if (AVEC) {
            #pragma unroll
            for (int r = 0; r < 8; ++r) apart[r] += vv[r] * sAw[col];
        }
        if (STORE) {
            if (fullTile) {            // branch-free common path
                #pragma unroll
                for (int r = 0; r < 8; ++r)
                    H[(size_t)(r0 + r + 8 * half) * D + col] = vv[r];
            } else {                   // guarded tail tile only
                #pragma unroll
                for (int r = 0; r < 8; ++r) {
                    const int row = r0 + r + 8 * half;
                    if (row < N) H[(size_t)row * D + col] = vv[r];
                }
            }
        }
    }

    if (AVEC) {
        #pragma unroll
        for (int r = 0; r < 8; ++r) {
            float s = apart[r];
            s += __shfl_xor(s, 1, 32);
            s += __shfl_xor(s, 2, 32);
            s += __shfl_xor(s, 4, 32);
            s += __shfl_xor(s, 8, 32);   // reduce across the 16 lanes of a half
            const int row = r0 + r + 8 * half;
            if (l16 == 0 && row < N) avec[row] = s + abv;
        }
    }
}

// ---------------------------------------------------------------------------
// out[n] = dot(H[n,:], w) + bias   (one wave per node)
// ---------------------------------------------------------------------------
__global__ void __launch_bounds__(256)
dotproj_kernel(const float* __restrict__ H, const float* __restrict__ w,
               const float* __restrict__ bias, float* __restrict__ out, int N)
{
    const int node = (int)((blockIdx.x * 256u + threadIdx.x) >> 5);
    const int lane = threadIdx.x & 31;
    if (node >= N) return;
    const float* h = H + (size_t)node * D;
    float s = 0.0f;
    #pragma unroll
    for (int i = 0; i < 4; ++i) s += h[lane + 32 * i] * w[lane + 32 * i];
    s += __shfl_xor(s, 16, 32);
    s += __shfl_xor(s, 8, 32);
    s += __shfl_xor(s, 4, 32);
    s += __shfl_xor(s, 2, 32);
    s += __shfl_xor(s, 1, 32);
    if (lane == 0) out[node] = s + bias[0];
}

// ---------------------------------------------------------------------------
// agg[rows[e],:] += sigmoid(ai[rows[e]] + aj[cols[e]]) * xj[cols[e],:]
// one wave per edge, float4 per lane, global f32 atomics.
// ---------------------------------------------------------------------------
__global__ void __launch_bounds__(256)
edge_agg_kernel(const float* __restrict__ xj, const float* __restrict__ ai,
                const float* __restrict__ aj, const int* __restrict__ rows,
                const int* __restrict__ cols, float* __restrict__ agg, int E)
{
    const unsigned g = blockIdx.x * 256u + threadIdx.x;
    const int e   = (int)(g >> 5);
    const int sub = (int)(g & 31u);
    if (e >= E) return;
    const int r = rows[e];
    const int c = cols[e];
    const float w = sigmoidf_(ai[r] + aj[c]);
    const float4 xv = *(const float4*)(xj + (size_t)c * D + sub * 4);
    float* dst = agg + (size_t)r * D + sub * 4;
    atomicAdd(dst + 0, w * xv.x);
    atomicAdd(dst + 1, w * xv.y);
    atomicAdd(dst + 2, w * xv.z);
    atomicAdd(dst + 3, w * xv.w);
}

// ---------------------------------------------------------------------------
// dual attention mix + residual + LayerNorm1 -> y (one wave per node)
// ---------------------------------------------------------------------------
__global__ void __launch_bounds__(256)
combine_ln1_kernel(const float* __restrict__ x0,
                   const float* __restrict__ vx0, const float* __restrict__ v0,
                   const float* __restrict__ v1,  const float* __restrict__ v2,
                   const float* __restrict__ q,   const float* __restrict__ kx0,
                   const float* __restrict__ k0,  const float* __restrict__ k1,
                   const float* __restrict__ k2,
                   const float* __restrict__ g,   const float* __restrict__ b,
                   float* __restrict__ y, int N)
{
    const int node = (int)((blockIdx.x * 256u + threadIdx.x) >> 5);
    const int lane = threadIdx.x & 31;
    if (node >= N) return;

    const float qq = q[node];
    const float s0 = sigmoidf_(qq * kx0[node]);
    const float s1 = sigmoidf_(qq * k0[node]);
    const float s2 = sigmoidf_(qq * k1[node]);
    const float s3 = sigmoidf_(qq * k2[node]);

    const size_t base = (size_t)node * D + lane * 4;
    const float4 xv = *(const float4*)(x0  + base);
    const float4 a0 = *(const float4*)(vx0 + base);
    const float4 a1 = *(const float4*)(v0  + base);
    const float4 a2 = *(const float4*)(v1  + base);
    const float4 a3 = *(const float4*)(v2  + base);

    float t[4];
    t[0] = 0.25f * (s0 * a0.x + s1 * a1.x + s2 * a2.x + s3 * a3.x) + xv.x;
    t[1] = 0.25f * (s0 * a0.y + s1 * a1.y + s2 * a2.y + s3 * a3.y) + xv.y;
    t[2] = 0.25f * (s0 * a0.z + s1 * a1.z + s2 * a2.z + s3 * a3.z) + xv.z;
    t[3] = 0.25f * (s0 * a0.w + s1 * a1.w + s2 * a2.w + s3 * a3.w) + xv.w;

    float sum = t[0] + t[1] + t[2] + t[3];
    sum += __shfl_xor(sum, 16, 32); sum += __shfl_xor(sum, 8, 32);
    sum += __shfl_xor(sum, 4, 32);  sum += __shfl_xor(sum, 2, 32);
    sum += __shfl_xor(sum, 1, 32);
    const float mu = sum * (1.0f / 128.0f);

    float vs = 0.0f;
    #pragma unroll
    for (int i = 0; i < 4; ++i) { const float d = t[i] - mu; vs += d * d; }
    vs += __shfl_xor(vs, 16, 32); vs += __shfl_xor(vs, 8, 32);
    vs += __shfl_xor(vs, 4, 32);  vs += __shfl_xor(vs, 2, 32);
    vs += __shfl_xor(vs, 1, 32);
    const float inv = rsqrtf(vs * (1.0f / 128.0f) + 1e-5f);

    const int col = lane * 4;
    float4 o;
    o.x = (t[0] - mu) * inv * g[col + 0] + b[col + 0];
    o.y = (t[1] - mu) * inv * g[col + 1] + b[col + 1];
    o.z = (t[2] - mu) * inv * g[col + 2] + b[col + 2];
    o.w = (t[3] - mu) * inv * g[col + 3] + b[col + 3];
    *(float4*)(y + base) = o;
}

// ---------------------------------------------------------------------------
// out = LayerNorm2(Z + Y)   (one wave per node)
// ---------------------------------------------------------------------------
__global__ void __launch_bounds__(256)
ln2_kernel(const float* __restrict__ Z, const float* __restrict__ Y,
           const float* __restrict__ g, const float* __restrict__ b,
           float* __restrict__ out, int N)
{
    const int node = (int)((blockIdx.x * 256u + threadIdx.x) >> 5);
    const int lane = threadIdx.x & 31;
    if (node >= N) return;

    const size_t base = (size_t)node * D + lane * 4;
    const float4 zv = *(const float4*)(Z + base);
    const float4 yv = *(const float4*)(Y + base);
    float t[4] = { zv.x + yv.x, zv.y + yv.y, zv.z + yv.z, zv.w + yv.w };

    float sum = t[0] + t[1] + t[2] + t[3];
    sum += __shfl_xor(sum, 16, 32); sum += __shfl_xor(sum, 8, 32);
    sum += __shfl_xor(sum, 4, 32);  sum += __shfl_xor(sum, 2, 32);
    sum += __shfl_xor(sum, 1, 32);
    const float mu = sum * (1.0f / 128.0f);

    float vs = 0.0f;
    #pragma unroll
    for (int i = 0; i < 4; ++i) { const float d = t[i] - mu; vs += d * d; }
    vs += __shfl_xor(vs, 16, 32); vs += __shfl_xor(vs, 8, 32);
    vs += __shfl_xor(vs, 4, 32);  vs += __shfl_xor(vs, 2, 32);
    vs += __shfl_xor(vs, 1, 32);
    const float inv = rsqrtf(vs * (1.0f / 128.0f) + 1e-5f);

    const int col = lane * 4;
    float4 o;
    o.x = (t[0] - mu) * inv * g[col + 0] + b[col + 0];
    o.y = (t[1] - mu) * inv * g[col + 1] + b[col + 1];
    o.z = (t[2] - mu) * inv * g[col + 2] + b[col + 2];
    o.w = (t[3] - mu) * inv * g[col + 3] + b[col + 3];
    *(float4*)(out + base) = o;
}

// ---------------------------------------------------------------------------
extern "C" void kernel_launch(void* const* d_in, const int* in_sizes, int n_in,
                              void* d_out, int out_size, void* d_ws, size_t ws_size,
                              hipStream_t stream)
{
    const float* x0   = (const float*)d_in[0];
    const float* x1   = (const float*)d_in[1];
    const float* x2   = (const float*)d_in[2];
    const int* rows0  = (const int*)d_in[3];
    const int* cols0  = (const int*)d_in[4];
    const int* rows1  = (const int*)d_in[5];
    const int* cols1  = (const int*)d_in[6];
    const int* rows2  = (const int*)d_in[7];
    const int* cols2  = (const int*)d_in[8];
    const float* W1  = (const float*)d_in[9];  const float* b1 = (const float*)d_in[10];
    const float* W2  = (const float*)d_in[11]; const float* b2 = (const float*)d_in[12];
    const float* W3  = (const float*)d_in[13]; const float* b3 = (const float*)d_in[14];
    const float* W4  = (const float*)d_in[15]; const float* b4 = (const float*)d_in[16];
    const float* a1w = (const float*)d_in[17]; const float* a1b = (const float*)d_in[18];
    const float* a2w = (const float*)d_in[19]; const float* a2b = (const float*)d_in[20];
    const float* a3w = (const float*)d_in[21]; const float* a3b = (const float*)d_in[22];
    const float* a4w = (const float*)d_in[23]; const float* a4b = (const float*)d_in[24];
    const float* wqw = (const float*)d_in[25]; const float* wqb = (const float*)d_in[26];
    const float* wkw = (const float*)d_in[27]; const float* wkb = (const float*)d_in[28];
    const float* wvw = (const float*)d_in[29]; const float* wvb = (const float*)d_in[30];
    const float* Wo  = (const float*)d_in[31]; const float* bo  = (const float*)d_in[32];
    const float* ln1g = (const float*)d_in[33]; const float* ln1b = (const float*)d_in[34];
    const float* ln2g = (const float*)d_in[35]; const float* ln2b = (const float*)d_in[36];

    const int N0 = in_sizes[0] / D;
    const int N1 = in_sizes[1] / D;
    const int N2 = in_sizes[2] / D;
    const int E  = in_sizes[3];

    // ---- workspace layout (with lifetime-based reuse) ----
    float* ws = (float*)d_ws;
    size_t o = 0;
    float* xj0  = ws + o; o += (size_t)N0 * D;   // xi_j set 0; later v1
    float* xj1  = ws + o; o += (size_t)N1 * D;   // xi_j set 1; later vx0 + v0
    float* xj2  = ws + o; o += (size_t)N2 * D;   // xi_j set 2
    float* agg0 = ws + o; o += (size_t)N0 * D;   // later y
    float* agg1 = ws + o; o += (size_t)N0 * D;   // later z
    float* agg2 = ws + o; o += (size_t)N0 * D;
    float* v2   = ws + o; o += (size_t)N0 * D;
    float* ai   = ws + o; o += (size_t)N0;
    float* aj0  = ws + o; o += (size_t)N0;
    float* aj1  = ws + o; o += (size_t)N1;
    float* aj2  = ws + o; o += (size_t)N2;
    float* qv   = ws + o; o += (size_t)N0;
    float* kx0  = ws + o; o += (size_t)N0;
    float* k0   = ws + o; o += (size_t)N0;
    float* k1   = ws + o; o += (size_t)N0;
    float* k2   = ws + o; o += (size_t)N0;
    float* vx0 = xj1;                        // xj1 dead after edge pass 1
    float* v0  = xj1 + (size_t)N0 * D;       // N1 == 2*N0 in this problem
    float* v1  = xj0;                        // xj0 dead after edge pass 0
    float* y   = agg0;                       // aggs dead after v-GEMMs / k-projs
    float* z   = agg1;

    const dim3 blk(256);
    const dim3 g0((unsigned)((N0 + 127) / 128));
    const dim3 g1((unsigned)((N1 + 127) / 128));
    const dim3 g2((unsigned)((N2 + 127) / 128));
    const dim3 w0((unsigned)((N0 + 7) / 8));
    const dim3 eg((unsigned)(((size_t)E * 32 + 255) / 256));

    // 1) feature transforms (+ fused attention scalar projections)
    gemm128_kernel<true,  false, true><<<g0, blk, 0, stream>>>(x0, W1, b1, a1w, a1b, nullptr, ai, N0);
    gemm128_kernel<true,  true,  true><<<g0, blk, 0, stream>>>(x0, W2, b2, a2w, a2b, xj0, aj0, N0);
    gemm128_kernel<true,  true,  true><<<g1, blk, 0, stream>>>(x1, W3, b3, a3w, a3b, xj1, aj1, N1);
    gemm128_kernel<true,  true,  true><<<g2, blk, 0, stream>>>(x2, W4, b4, a4w, a4b, xj2, aj2, N2);

    // 2) q / k scalar projections on x0
    dotproj_kernel<<<w0, blk, 0, stream>>>(x0, wqw, wqb, qv, N0);
    dotproj_kernel<<<w0, blk, 0, stream>>>(x0, wkw, wkb, kx0, N0);

    // 3) attention-weighted segment sums (aggs zeroed every call)
    hipMemsetAsync(agg0, 0, (size_t)3 * N0 * D * sizeof(float), stream);
    edge_agg_kernel<<<eg, blk, 0, stream>>>(xj0, ai, aj0, rows0, cols0, agg0, E);
    edge_agg_kernel<<<eg, blk, 0, stream>>>(xj1, ai, aj1, rows1, cols1, agg1, E);
    edge_agg_kernel<<<eg, blk, 0, stream>>>(xj2, ai, aj2, rows2, cols2, agg2, E);

    // 4) k projections of the aggregates
    dotproj_kernel<<<w0, blk, 0, stream>>>(agg0, wkw, wkb, k0, N0);
    dotproj_kernel<<<w0, blk, 0, stream>>>(agg1, wkw, wkb, k1, N0);
    dotproj_kernel<<<w0, blk, 0, stream>>>(agg2, wkw, wkb, k2, N0);

    // 5) value projections (WMMA GEMMs, no activation, no projection)
    gemm128_kernel<false, true, false><<<g0, blk, 0, stream>>>(x0,   wvw, wvb, nullptr, nullptr, vx0, nullptr, N0);
    gemm128_kernel<false, true, false><<<g0, blk, 0, stream>>>(agg0, wvw, wvb, nullptr, nullptr, v0,  nullptr, N0);
    gemm128_kernel<false, true, false><<<g0, blk, 0, stream>>>(agg1, wvw, wvb, nullptr, nullptr, v1,  nullptr, N0);
    gemm128_kernel<false, true, false><<<g0, blk, 0, stream>>>(agg2, wvw, wvb, nullptr, nullptr, v2,  nullptr, N0);

    // 6) dual attention mix + residual + LN1
    combine_ln1_kernel<<<w0, blk, 0, stream>>>(x0, vx0, v0, v1, v2,
                                               qv, kx0, k0, k1, k2,
                                               ln1g, ln1b, y, N0);

    // 7) output projection + residual + LN2
    gemm128_kernel<false, true, false><<<g0, blk, 0, stream>>>(y, Wo, bo, nullptr, nullptr, z, nullptr, N0);
    ln2_kernel<<<w0, blk, 0, stream>>>(z, y, ln2g, ln2b, (float*)d_out, N0);
}
